// Attention_20504173871523
// MI455X (gfx1250) — 8-bit matrix operand order repaired and checked by reading, compiled, not run on hardware
//
#include <hip/hip_runtime.h>
#include <hip/hip_bf16.h>

#define EPS 1e-5f
#define Bn   16
#define Nn   1025
#define Cn   768
#define Hn   12
#define Dn   64
#define On   (3*Cn)
#define Mn   (Bn*Nn)
#define BHn  (Bn*Hn)
#define NPAD 1088
#define ITIL 65
#define JTIL 68
#define DTIL 4
#define QCNT (BHn*Nn*Dn)
#define ACNT (BHn*Nn*Nn)

typedef __attribute__((ext_vector_type(16))) __bf16 v16bf;
typedef __attribute__((ext_vector_type(8)))  __bf16 v8bf;
typedef __attribute__((ext_vector_type(8)))  float  v8f;
typedef __attribute__((ext_vector_type(8)))  int    v8i;
typedef unsigned long long u64;

#define SC_ABSW 0
#define SC_GQ   2
#define SC_GK   3
#define SC_GV   4
#define SC_ABSA 5

static __device__ __forceinline__ float wave_sum_f(float v) {
  #pragma unroll
  for (int m = 16; m >= 1; m >>= 1) v += __shfl_xor(v, m, 32);
  return v;
}
static __device__ __forceinline__ int wave_sum_i(int v) {
  #pragma unroll
  for (int m = 16; m >= 1; m >>= 1) v += __shfl_xor(v, m, 32);
  return v;
}

__global__ void k_init(float* scal) {
  if (threadIdx.x < 64) scal[threadIdx.x] = 0.f;
}

__global__ void k_absw(const float* __restrict__ w, float* __restrict__ scal) {
  const int n = On * Cn;
  int stride = gridDim.x * blockDim.x;
  float s = 0.f;
  for (int i = blockIdx.x * blockDim.x + threadIdx.x; i < n; i += stride)
    s += fabsf(w[i]);
  s = wave_sum_f(s);
  if ((threadIdx.x & 31) == 0) atomicAdd(&scal[SC_ABSW], s);
}

__global__ void k_wq(const float* __restrict__ w, const float* __restrict__ scal,
                     __bf16* __restrict__ wq) {
  int i = blockIdx.x * blockDim.x + threadIdx.x;
  if (i >= On * Cn) return;
  float s = scal[SC_ABSW] / (float)(On * Cn) + EPS;
  float u = w[i] / s;
  wq[i] = (__bf16)rintf(fminf(fmaxf(u, -1.f), 1.f));
}

__global__ void k_xcvt(const float* __restrict__ x, __bf16* __restrict__ xb) {
  int i = blockIdx.x * blockDim.x + threadIdx.x;
  if (i >= Mn * Cn) return;
  xb[i] = (__bf16)x[i];
}

__global__ void k_gemm_qkv(const __bf16* __restrict__ xb, const __bf16* __restrict__ wq,
                           const float* __restrict__ scal, float* __restrict__ qkv) {
  const int MT = Mn / 16, NT = On / 16;
  int wave = (int)((blockIdx.x * blockDim.x + threadIdx.x) >> 5);
  if (wave >= MT * NT) return;
  int lane = threadIdx.x & 31;
  int mt = wave % MT, ot = wave / MT;
  int kb = (lane < 16) ? 0 : 8;
  const __bf16* pa = xb + (size_t)(mt * 16 + (lane & 15)) * Cn + kb;
  const __bf16* pb = wq + (size_t)(ot * 16 + (lane & 15)) * Cn + kb;
  v8f acc; for (int t = 0; t < 8; t++) acc[t] = 0.f;
  for (int k0 = 0; k0 < Cn; k0 += 32) {
    v8bf a0 = *(const v8bf*)(pa + k0);
    v8bf a1 = *(const v8bf*)(pa + k0 + 16);
    v8bf b0 = *(const v8bf*)(pb + k0);
    v8bf b1 = *(const v8bf*)(pb + k0 + 16);
    __builtin_prefetch(pa + k0 + 32, 0, 1);
    __builtin_prefetch(pb + k0 + 32, 0, 1);
    v16bf av, bv;
    #pragma unroll
    for (int t = 0; t < 8; t++) { av[t] = a0[t]; av[8 + t] = a1[t];
                                  bv[t] = b0[t]; bv[8 + t] = b1[t]; }
    acc = __builtin_amdgcn_wmma_f32_16x16x32_bf16(false, av, false, bv,
                                                  (short)0, acc, false, false);
  }
  float sw = scal[SC_ABSW] / (float)(On * Cn) + EPS;
  int col = lane & 15;
  int rbs = (lane >= 16) ? 8 : 0;
  #pragma unroll
  for (int r = 0; r < 8; r++)
    qkv[(size_t)(mt * 16 + r + rbs) * On + ot * 16 + col] = acc[r] * sw;
}

__global__ void k_rowsum(const float* __restrict__ qkv, float* __restrict__ rsum,
                         float* __restrict__ scal) {
  int gw = (int)((blockIdx.x * blockDim.x + threadIdx.x) >> 5);
  if (gw >= 3 * BHn * Nn) return;
  int lane = threadIdx.x & 31;
  int n = gw % Nn; int t = gw / Nn;
  int h = t % Hn;  t /= Hn;
  int b = t % Bn;  int s = t / Bn;
  const float* p = qkv + (size_t)(b * Nn + n) * On + (s * Hn + h) * Dn;
  float v = fabsf(p[2 * lane]) + fabsf(p[2 * lane + 1]);
  v = wave_sum_f(v);
  if (lane == 0) {
    rsum[gw] = v;
    atomicAdd(&scal[SC_GQ + s], v / (v + EPS));
  }
}

__global__ void k_quant(const float* __restrict__ qkv, const float* __restrict__ rsum,
                        const float* __restrict__ scal,
                        signed char* __restrict__ sq, signed char* __restrict__ sk,
                        signed char* __restrict__ vT) {
  size_t idx = (size_t)blockIdx.x * blockDim.x + threadIdx.x;
  if (idx >= (size_t)3 * QCNT) return;
  int d = (int)(idx % Dn); size_t r = idx / Dn;
  int n = (int)(r % Nn); r /= Nn;
  int h = (int)(r % Hn); r /= Hn;
  int b = (int)(r % Bn); int s = (int)(r / Bn);
  float rs = rsum[(((size_t)s * Bn + b) * Hn + h) * Nn + n];
  float st = scal[SC_GQ + s] / (float)QCNT + EPS;
  float val = qkv[(size_t)(b * Nn + n) * On + (s * Hn + h) * Dn + d];
  float u = (val / (rs + EPS)) / st;
  signed char q = (signed char)(int)rintf(fminf(fmaxf(u, -1.f), 1.f));
  size_t qi = (((size_t)b * Hn + h) * Nn + n) * Dn + d;
  if (s == 0)      sq[qi] = q;
  else if (s == 1) sk[qi] = q;
  else             vT[(((size_t)b * Hn + h) * Dn + d) * NPAD + n] = q;
}

__global__ void k_padv(signed char* __restrict__ vT) {
  int total = BHn * Dn * (NPAD - Nn);
  int idx = blockIdx.x * blockDim.x + threadIdx.x;
  if (idx >= total) return;
  int n = Nn + idx % (NPAD - Nn);
  int rd = idx / (NPAD - Nn);
  vT[(size_t)rd * NPAD + n] = 0;
}

__global__ void k_gemm_attn(const signed char* __restrict__ sq,
                            const signed char* __restrict__ sk,
                            signed char* __restrict__ attn, float* __restrict__ scal) {
  int wave = (int)((blockIdx.x * blockDim.x + threadIdx.x) >> 5);
  if (wave >= BHn * ITIL * JTIL) return;
  int lane = threadIdx.x & 31;
  int jt = wave % JTIL; int t = wave / JTIL;
  int it = t % ITIL;    int bh = t / ITIL;
  int i0 = it * 16, j0 = jt * 16;
  int rowi = i0 + (lane & 15);
  int rowj = j0 + (lane & 15);
  v8i a, bm;
  if (rowi < Nn) {
    const signed char* pa = sq + ((size_t)bh * Nn + rowi) * Dn + ((lane < 16) ? 0 : 16);
    u64 c0 = *(const u64*)(pa);      u64 c1 = *(const u64*)(pa + 8);
    u64 c2 = *(const u64*)(pa + 32); u64 c3 = *(const u64*)(pa + 40);
    a[0] = (int)(unsigned)c0; a[1] = (int)(unsigned)(c0 >> 32);
    a[2] = (int)(unsigned)c1; a[3] = (int)(unsigned)(c1 >> 32);
    a[4] = (int)(unsigned)c2; a[5] = (int)(unsigned)(c2 >> 32);
    a[6] = (int)(unsigned)c3; a[7] = (int)(unsigned)(c3 >> 32);
  } else { for (int u = 0; u < 8; u++) a[u] = 0; }
  if (rowj < Nn) {
    const signed char* pb = sk + ((size_t)bh * Nn + rowj) * Dn + ((lane < 16) ? 0 : 16);
    int4 q0 = *(const int4*)(pb);
    int4 q1 = *(const int4*)(pb + 32);
    bm[0] = q0.x; bm[1] = q0.y; bm[2] = q0.z; bm[3] = q0.w;
    bm[4] = q1.x; bm[5] = q1.y; bm[6] = q1.z; bm[7] = q1.w;
  } else { for (int u = 0; u < 8; u++) bm[u] = 0; }
  v8i cz; for (int u = 0; u < 8; u++) cz[u] = 0;
  v8i c = __builtin_amdgcn_wmma_i32_16x16x64_iu8(true, a, true, bm, cz, false, false);
  int col = lane & 15;
  int rbs = (lane >= 16) ? 8 : 0;
  int asum = 0;
  #pragma unroll
  for (int r = 0; r < 8; r++) {
    int i = i0 + r + rbs;
    if (i < Nn) {
      attn[((size_t)bh * Nn + i) * NPAD + j0 + col] = (signed char)c[r];
      if (j0 + col < Nn) { int v = c[r]; asum += (v < 0) ? -v : v; }
    }
  }
  asum = wave_sum_i(asum);
  if (lane == 0) atomicAdd(&scal[SC_ABSA], (float)asum);
}

__global__ void k_gemm_out(const signed char* __restrict__ attn,
                           const signed char* __restrict__ vT,
                           const float* __restrict__ scal, float* __restrict__ out) {
  int wave = (int)((blockIdx.x * blockDim.x + threadIdx.x) >> 5);
  if (wave >= BHn * ITIL * DTIL) return;
  int lane = threadIdx.x & 31;
  int dt = wave % DTIL; int t = wave / DTIL;
  int it = t % ITIL;    int bh = t / ITIL;
  int i0 = it * 16, d0 = dt * 16;
  int rowi = i0 + (lane & 15);
  int coln = d0 + (lane & 15);
  float qs = scal[SC_GQ] / (float)QCNT + EPS;
  float ks = scal[SC_GK] / (float)QCNT + EPS;
  float vs = scal[SC_GV] / (float)QCNT + EPS;
  float cf = 0.125f * qs * ks;
  float qk_s = cf * (scal[SC_ABSA] / (float)ACNT) + EPS;
  float inv_t = cf / qk_s;
  const signed char* pa0 = attn + ((size_t)bh * Nn + ((rowi < Nn) ? rowi : 0)) * NPAD
                                + ((lane < 16) ? 0 : 16);
  const signed char* pv0 = vT + ((size_t)bh * Dn + coln) * NPAD + ((lane < 16) ? 0 : 16);
  bool avalid = (rowi < Nn);
  v8i acc; for (int u = 0; u < 8; u++) acc[u] = 0;
  for (int k0 = 0; k0 < NPAD; k0 += 64) {
    v8i a;
    if (avalid) {
      #pragma unroll
      for (int ch = 0; ch < 4; ch++) {
        u64 u = *(const u64*)(pa0 + k0 + (ch & 1) * 8 + (ch >> 1) * 32);
        unsigned lo = 0, hi = 0;
        #pragma unroll
        for (int tb = 0; tb < 8; tb++) {
          int raw = (int)(signed char)(unsigned char)(u >> (8 * tb));
          float f = fminf(fmaxf((float)raw * inv_t, -1.f), 1.f);
          unsigned byte = (unsigned)((int)rintf(f) & 0xFF);
          if (tb < 4) lo |= byte << (8 * tb); else hi |= byte << (8 * (tb - 4));
        }
        a[2 * ch] = (int)lo; a[2 * ch + 1] = (int)hi;
      }
    } else { for (int u = 0; u < 8; u++) a[u] = 0; }
    int4 q0 = *(const int4*)(pv0 + k0);
    int4 q1 = *(const int4*)(pv0 + k0 + 32);
    __builtin_prefetch(pv0 + k0 + 64, 0, 1);
    v8i bm;
    bm[0] = q0.x; bm[1] = q0.y; bm[2] = q0.z; bm[3] = q0.w;
    bm[4] = q1.x; bm[5] = q1.y; bm[6] = q1.z; bm[7] = q1.w;
    acc = __builtin_amdgcn_wmma_i32_16x16x64_iu8(true, a, true, bm, acc, false, false);
  }
  float of = qk_s * vs;
  int b = bh / Hn, h = bh % Hn;
  int col = lane & 15;
  int rbs = (lane >= 16) ? 8 : 0;
  #pragma unroll
  for (int r = 0; r < 8; r++) {
    int i = i0 + r + rbs;
    if (i < Nn)
      out[((size_t)b * Nn + i) * Cn + h * Dn + d0 + col] = (float)acc[r] * of;
  }
}

extern "C" void kernel_launch(void* const* d_in, const int* in_sizes, int n_in,
                              void* d_out, int out_size, void* d_ws, size_t ws_size,
                              hipStream_t stream) {
  (void)in_sizes; (void)n_in; (void)out_size; (void)ws_size;
  const float* x = (const float*)d_in[0];
  const float* w = (const float*)d_in[1];
  float* out = (float*)d_out;
  char* ws = (char*)d_ws;
  size_t off = 0;
  float* scal = (float*)(ws + off);          off += 256;
  __bf16* wq  = (__bf16*)(ws + off);         off += (size_t)On * Cn * 2;
  __bf16* xb  = (__bf16*)(ws + off);         off += (size_t)Mn * Cn * 2;
  float* qkv  = (float*)(ws + off);          off += (size_t)Mn * On * 4;
  float* rsum = (float*)(ws + off);          off += (size_t)3 * BHn * Nn * 4;
  signed char* sq   = (signed char*)(ws + off); off += (size_t)BHn * Nn * Dn;
  signed char* sk   = (signed char*)(ws + off); off += (size_t)BHn * Nn * Dn;
  signed char* vT   = (signed char*)(ws + off); off += (size_t)BHn * Dn * NPAD;
  signed char* attn = (signed char*)(ws + off); off += (size_t)BHn * Nn * NPAD;

  k_init<<<1, 64, 0, stream>>>(scal);
  k_absw<<<512, 256, 0, stream>>>(w, scal);
  k_wq<<<(On * Cn + 255) / 256, 256, 0, stream>>>(w, scal, wq);
  k_xcvt<<<(Mn * Cn + 255) / 256, 256, 0, stream>>>(x, xb);
  k_gemm_qkv<<<((Mn / 16) * (On / 16) + 7) / 8, 256, 0, stream>>>(xb, wq, scal, qkv);
  k_rowsum<<<(3 * BHn * Nn + 7) / 8, 256, 0, stream>>>(qkv, rsum, scal);
  {
    size_t nq = (size_t)3 * QCNT;
    k_quant<<<(unsigned)((nq + 255) / 256), 256, 0, stream>>>(qkv, rsum, scal, sq, sk, vT);
  }
  k_padv<<<(BHn * Dn * (NPAD - Nn) + 255) / 256, 256, 0, stream>>>(vT);
  k_gemm_attn<<<(BHn * ITIL * JTIL + 7) / 8, 256, 0, stream>>>(sq, sk, attn, scal);
  k_gemm_out<<<(BHn * ITIL * DTIL + 7) / 8, 256, 0, stream>>>(attn, vT, scal, out);
}
